// PhiMoESparseMoeBlock_2886218023363
// MI455X (gfx1250) — compile-verified, matrix operand order checked by reading, not run on hardware
//
#include <hip/hip_runtime.h>
#include <hip/hip_bf16.h>

typedef __attribute__((ext_vector_type(16))) _Float16 v16h;
typedef __attribute__((ext_vector_type(8)))  float    v8f;
typedef __attribute__((ext_vector_type(4)))  unsigned u32x4;
typedef __attribute__((ext_vector_type(8)))  int      i32x8;
typedef __attribute__((ext_vector_type(4)))  int      i32x4;

#define T 4096
#define D 1024
#define F 2048
#define E 8
#define JEPS 0.01f

#define MT 128   // row tile (tokens per tile)
#define SA 40    // A-stage LDS stride (halves): 32 + 8 pad (16B-aligned rows)
#define SBG 1032 // gate/up weight panel stride (halves): 1024 + 8 pad
#define SBD 2056 // down weight panel stride (halves): 2048 + 8 pad

#if defined(__has_builtin)
#if __has_builtin(__builtin_amdgcn_tensor_load_to_lds) && __has_builtin(__builtin_amdgcn_s_wait_tensorcnt)
#define USE_TDM 1
#warning CDNA5 probe: tensor_load_to_lds + s_wait_tensorcnt AVAILABLE (TDM path ON)
#if __has_include(<hip/amd_detail/amd_gfx1250_TDM.h>)
#define TDM_6ARG 1
#warning CDNA5 probe: gfx1250_TDM.h present -> 6-arg builtin form
#else
#warning CDNA5 probe: no TDM header -> 5-arg ROCm-7.2 builtin form
#endif
#else
#warning CDNA5 probe: tensor_load_to_lds builtin MISSING (manual staging fallback)
#endif
#endif

// ---------------------------------------------------------------------------
// Kernel 0: zero output accumulator + 16 routing counters
// ---------------------------------------------------------------------------
__global__ void k_zero(float* __restrict__ out, int n4, int* __restrict__ counts) {
    int i = blockIdx.x * blockDim.x + threadIdx.x;
    if (i < n4) ((float4*)out)[i] = float4{0.f, 0.f, 0.f, 0.f};
    if (blockIdx.x == 0 && threadIdx.x < 16) counts[threadIdx.x] = 0;
}

// ---------------------------------------------------------------------------
// Kernel 0b: pre-convert x to f16 once
// ---------------------------------------------------------------------------
__global__ void k_cvtx(const float* __restrict__ x, _Float16* __restrict__ x16) {
    int i = blockIdx.x * blockDim.x + threadIdx.x;   // 8 floats per thread
    const float4* src = (const float4*)x + (size_t)i * 2;
    _Float16* dst = x16 + (size_t)i * 8;
    union { _Float16 h[4]; uint2 u; } t0, t1;
    float4 a = src[0], b = src[1];
    t0.h[0] = (_Float16)a.x; t0.h[1] = (_Float16)a.y;
    t0.h[2] = (_Float16)a.z; t0.h[3] = (_Float16)a.w;
    t1.h[0] = (_Float16)b.x; t1.h[1] = (_Float16)b.y;
    t1.h[2] = (_Float16)b.z; t1.h[3] = (_Float16)b.w;
    *(uint2*)dst = t0.u;
    *(uint2*)(dst + 4) = t1.u;
}

// ---------------------------------------------------------------------------
// Kernel 1: router logits, exact fp32
// ---------------------------------------------------------------------------
__global__ __launch_bounds__(256) void k_router(const float* __restrict__ x,
                                                const float* __restrict__ gw,
                                                float* __restrict__ logits) {
    int t = blockIdx.x;
    int e = threadIdx.x >> 5;
    int lane = threadIdx.x & 31;
    const float4* xr = (const float4*)(x + (size_t)t * D);
    const float4* gr = (const float4*)(gw + (size_t)e * D);
    float s = 0.f;
#pragma unroll
    for (int c = 0; c < 8; ++c) {
        float4 a = xr[lane * 8 + c];
        float4 b = gr[lane * 8 + c];
        s += a.x * b.x + a.y * b.y + a.z * b.z + a.w * b.w;
    }
    for (int off = 16; off > 0; off >>= 1) s += __shfl_down(s, off, 32);
    if (lane == 0) logits[t * E + e] = s;
}

// ---------------------------------------------------------------------------
// Kernel 2: sparsemixer top-2 (faithful port) + per-(expert,rank) token lists
// ---------------------------------------------------------------------------
__global__ void k_sparsemixer(const float* __restrict__ logits,
                              int* __restrict__ rows, float* __restrict__ wtsb,
                              int* __restrict__ counts) {
    int t = blockIdx.x * blockDim.x + threadIdx.x;
    if (t >= T) return;
    float s[E];
#pragma unroll
    for (int i = 0; i < E; ++i) s[i] = logits[t * E + i];

    float mx1 = s[0]; int i1 = 0;
#pragma unroll
    for (int i = 1; i < E; ++i) if (s[i] > mx1) { mx1 = s[i]; i1 = i; }
    float den1 = 0.f;
#pragma unroll
    for (int i = 0; i < E; ++i) {
        float fac = fmaxf(fabsf(s[i]), mx1);
        bool masked = (mx1 - s[i]) / fac > 2.f * JEPS;
        if (!masked) den1 += expf(s[i] - mx1);
    }
    float w1 = 1.f / den1;

    float mx2 = -__builtin_huge_valf(); int i2 = 0;
#pragma unroll
    for (int i = 0; i < E; ++i) if (i != i1 && s[i] > mx2) { mx2 = s[i]; i2 = i; }
    float den2 = 0.f;
#pragma unroll
    for (int i = 0; i < E; ++i) {
        if (i == i1) continue;
        float fac = fmaxf(fabsf(s[i]), mx2);
        bool masked = (mx2 - s[i]) / fac > 2.f * JEPS;
        if (!masked) den2 += expf(s[i] - mx2);
    }
    float w2 = 1.f / den2;

    int p1 = atomicAdd(&counts[2 * i1 + 0], 1);
    rows[(2 * i1 + 0) * T + p1] = t;  wtsb[(2 * i1 + 0) * T + p1] = w1;
    int p2 = atomicAdd(&counts[2 * i2 + 1], 1);
    rows[(2 * i2 + 1) * T + p2] = t;  wtsb[(2 * i2 + 1) * T + p2] = w2;
}

// ---------------------------------------------------------------------------
// Kernel 3: prefix over segment counts
// ---------------------------------------------------------------------------
__global__ void k_prefix(const int* __restrict__ counts, int* __restrict__ basep) {
    if (threadIdx.x == 0) {
        int a = 0;
        for (int i = 0; i < 16; ++i) { basep[i] = a; a += counts[i]; }
        basep[16] = a;
    }
}

// ---------------------------------------------------------------------------
// WMMA fragment loader (wave32, ISA 7.12.2): two aligned uint4 -> ds_load_b128
// ---------------------------------------------------------------------------
__device__ __forceinline__ v16h load_frag16(const _Float16* p, int stride, int lane) {
    union { v16h h; uint4 q[2]; } f;
    const _Float16* r = p + (lane & 15) * stride + ((lane >> 4) & 1) * 8;
    f.q[0] = *(const uint4*)(r);
    f.q[1] = *(const uint4*)(r + 16);
    return f.h;
}

__device__ __forceinline__ void st4h(_Float16* dst, float4 v) {
    union { _Float16 h[4]; uint2 u; } t;
    t.h[0] = (_Float16)v.x; t.h[1] = (_Float16)v.y;
    t.h[2] = (_Float16)v.z; t.h[3] = (_Float16)v.w;
    *(uint2*)dst = t.u;
}

#ifdef USE_TDM
// ---------------------------------------------------------------------------
// TDM: async-load one (rows x 32) f16 tile (rows contiguous, stride F) into
// LDS with 4-DWORD pad per 16-DWORD row => padded stride SA=40 halves.
// ---------------------------------------------------------------------------
__device__ __forceinline__ void tdm_tile_f16(unsigned lds_addr,
                                             const _Float16* gsrc, int nrows) {
    unsigned long long ga = (unsigned long long)(size_t)gsrc;
    u32x4 g0;
    g0[0] = 1u;                                        // count=1, user D#
    g0[1] = lds_addr;                                  // LDS byte address
    g0[2] = (unsigned)(ga & 0xffffffffu);              // global_addr[31:0]
    g0[3] = (unsigned)((ga >> 32) & 0x01ffffffu) | (2u << 30);  // [56:32]|type=2
    i32x8 g1;
    g1[0] = (1 << 16) | (1 << 20) | (3 << 22) | (3 << 25); // dsz=2B,pad,int16DW,amt4DW
    g1[1] = (int)((unsigned)F << 16);                  // tensor_dim0 = F (lo16)
    g1[2] = (int)((unsigned)(2 * T) << 16);            // dim0 hi=0 | tensor_dim1 lo
    g1[3] = (int)(32u << 16);                          // dim1 hi=0 | tile_dim0=32
    g1[4] = nrows;                                     // tile_dim1, tile_dim2=0
    g1[5] = F;                                         // tensor_dim0_stride lo32
    g1[6] = 0;
    g1[7] = 0;
    i32x4 z4 = {0, 0, 0, 0};
#ifdef TDM_6ARG
    i32x8 z8 = {0, 0, 0, 0, 0, 0, 0, 0};
    __builtin_amdgcn_tensor_load_to_lds(g0, g1, z4, z4, z8, 0);
#else
    __builtin_amdgcn_tensor_load_to_lds(g0, g1, z4, z4, 0);
#endif
}
#endif

// ---------------------------------------------------------------------------
// Kernel 4: gathered gate/up GEMM + SiLU. Weight panels RESIDENT in LDS,
// row tile MT=128, waves 4(m) x 2(n): 6 frag loads per 8 WMMAs.
// ---------------------------------------------------------------------------
__global__ __launch_bounds__(256) void k_gateup(const _Float16* __restrict__ x16,
        const float* __restrict__ wg, const float* __restrict__ wu,
        const int* __restrict__ rows, const int* __restrict__ counts,
        const int* __restrict__ basep, _Float16* __restrict__ hbuf) {
    int f0  = blockIdx.x * 64;
    int seg = blockIdx.y;
    int e   = seg >> 1;
    int cnt = counts[seg];
    if (cnt == 0) return;
    int hb  = basep[seg];

    extern __shared__ char smem[];
    _Float16* Bg  = (_Float16*)smem;                 // 64 x SBG
    _Float16* Bu  = Bg + 64 * SBG;                   // 64 x SBG
    _Float16* Ast = Bu + 64 * SBG;                   // 2 x MT x SA
    int*      toks = (int*)(Ast + 2 * MT * SA);      // MT

    int tid = threadIdx.x;
    const float* wgp = wg + (size_t)e * F * D;
    const float* wup = wu + (size_t)e * F * D;

    for (int c = tid; c < 2 * 64 * 256; c += 256) {
        int mat = c >> 14;
        int rem = c & 16383;
        int n = rem >> 8, q = rem & 255;
        const float* src = (mat ? wup : wgp) + (size_t)(f0 + n) * D + q * 4;
        _Float16* dst = (mat ? Bu : Bg) + n * SBG + q * 4;
        st4h(dst, *(const float4*)src);
    }

    int wave = tid >> 5, lane = tid & 31;
    int wm = wave >> 1, wn = wave & 1;               // 4 x 2
    int am0 = tid >> 2, aq = (tid & 3) * 8;          // A chunks: rows 0..63
    int am1 = am0 + 64;                              //           rows 64..127
    int half8 = ((lane >> 4) & 1) * 8;
    int ncol = lane & 15;

    for (int r0 = 0; r0 < cnt; r0 += MT) {
        __syncthreads();
        if (tid < MT) {
            int r = r0 + tid;
            toks[tid] = rows[seg * T + (r < cnt ? r : cnt - 1)];
        }
        __syncthreads();

        const _Float16* xr0 = x16 + (size_t)toks[am0] * D + aq;
        const _Float16* xr1 = x16 + (size_t)toks[am1] * D + aq;
        uint4 av0 = *(const uint4*)(xr0);
        uint4 av1 = *(const uint4*)(xr1);
        *(uint4*)&Ast[am0 * SA + aq] = av0;
        *(uint4*)&Ast[am1 * SA + aq] = av1;

        v8f accg[2][2] = {}, accu[2][2] = {};
        int stage = 0;
#pragma unroll 1
        for (int it = 0; it < D / 32; ++it) {
            __syncthreads();
            int kn = (it + 1) * 32;
            if (kn < D) {
                av0 = *(const uint4*)(xr0 + kn);
                av1 = *(const uint4*)(xr1 + kn);
            }
            const _Float16* As = &Ast[stage * MT * SA];
            v16h a0 = load_frag16(&As[(wm * 32 + 0) * SA], SA, lane);
            v16h a1 = load_frag16(&As[(wm * 32 + 16) * SA], SA, lane);
            v16h bg0 = load_frag16(&Bg[(wn * 32 + 0) * SBG + it * 32], SBG, lane);
            v16h bg1 = load_frag16(&Bg[(wn * 32 + 16) * SBG + it * 32], SBG, lane);
            v16h bu0 = load_frag16(&Bu[(wn * 32 + 0) * SBG + it * 32], SBG, lane);
            v16h bu1 = load_frag16(&Bu[(wn * 32 + 16) * SBG + it * 32], SBG, lane);
            accg[0][0] = __builtin_amdgcn_wmma_f32_16x16x32_f16(false, a0, false, bg0, (short)0, accg[0][0], false, false);
            accg[1][0] = __builtin_amdgcn_wmma_f32_16x16x32_f16(false, a1, false, bg0, (short)0, accg[1][0], false, false);
            accg[0][1] = __builtin_amdgcn_wmma_f32_16x16x32_f16(false, a0, false, bg1, (short)0, accg[0][1], false, false);
            accg[1][1] = __builtin_amdgcn_wmma_f32_16x16x32_f16(false, a1, false, bg1, (short)0, accg[1][1], false, false);
            accu[0][0] = __builtin_amdgcn_wmma_f32_16x16x32_f16(false, a0, false, bu0, (short)0, accu[0][0], false, false);
            accu[1][0] = __builtin_amdgcn_wmma_f32_16x16x32_f16(false, a1, false, bu0, (short)0, accu[1][0], false, false);
            accu[0][1] = __builtin_amdgcn_wmma_f32_16x16x32_f16(false, a0, false, bu1, (short)0, accu[0][1], false, false);
            accu[1][1] = __builtin_amdgcn_wmma_f32_16x16x32_f16(false, a1, false, bu1, (short)0, accu[1][1], false, false);
            if (kn < D) {
                stage ^= 1;
                _Float16* Asw = &Ast[stage * MT * SA];
                *(uint4*)&Asw[am0 * SA + aq] = av0;
                *(uint4*)&Asw[am1 * SA + aq] = av1;
            }
        }

#pragma unroll
        for (int ni = 0; ni < 2; ++ni) {
            int gn = f0 + wn * 32 + ni * 16 + ncol;
#pragma unroll
            for (int mi = 0; mi < 2; ++mi)
#pragma unroll
                for (int v = 0; v < 8; ++v) {
                    int gm = wm * 32 + mi * 16 + half8 + v;
                    int r = r0 + gm;
                    if (r < cnt) {
                        float g = accg[mi][ni][v], u = accu[mi][ni][v];
                        float h = g / (1.f + expf(-g)) * u;
                        hbuf[(size_t)(hb + r) * F + gn] = (_Float16)h;
                    }
                }
        }
    }
}

// ---------------------------------------------------------------------------
// Kernel 5: down projection. Wd panel resident in LDS; h staged by TDM
// (128x32 tiles) when available. Row tile MT=128, waves 4(m) x 2(n):
// 4 frag loads per 4 WMMAs. Launched per rank -> deterministic +=.
// ---------------------------------------------------------------------------
__global__ __launch_bounds__(256) void k_down(const float* __restrict__ wd,
        const int* __restrict__ rows, const float* __restrict__ wtsb,
        const int* __restrict__ counts, const int* __restrict__ basep,
        const _Float16* __restrict__ hbuf, float* __restrict__ out, int rank) {
    int n0  = blockIdx.x * 64;
    int e   = blockIdx.y;
    int seg = 2 * e + rank;
    int cnt = counts[seg];
    if (cnt == 0) return;
    int hb  = basep[seg];

    extern __shared__ char smem[];
    _Float16* Bp  = (_Float16*)smem;                 // 64 x SBD
    _Float16* Ast = Bp + 64 * SBD;                   // 2 x MT x SA
    int*      toks = (int*)(Ast + 2 * MT * SA);      // MT
    float*    wsh  = (float*)(toks + MT);            // MT

    int tid = threadIdx.x;
    const float* wdp = wd + (size_t)e * D * F;

    for (int c = tid; c < 64 * 512; c += 256) {
        int n = c >> 9, q = (c & 511) * 4;
        st4h(&Bp[n * SBD + q], *(const float4*)(wdp + (size_t)(n0 + n) * F + q));
    }

    int wave = tid >> 5, lane = tid & 31;
    int wm = wave >> 1, wn = wave & 1;
    int half8 = ((lane >> 4) & 1) * 8;
    int ncol = lane & 15;
#ifdef USE_TDM
    unsigned ldsA0 = (unsigned)(size_t)&Ast[0];      // low32 of LDS-aperture VA
    unsigned ldsA1 = (unsigned)(size_t)&Ast[MT * SA];
#else
    int am0 = tid >> 2, aq = (tid & 3) * 8;
    int am1 = am0 + 64;
#endif

    for (int r0 = 0; r0 < cnt; r0 += MT) {
        __syncthreads();
        if (tid < MT) {
            int r = r0 + tid;
            bool ok = r < cnt;
            toks[tid] = ok ? rows[seg * T + r] : 0;
            wsh[tid]  = ok ? wtsb[seg * T + r] : 0.f;
        }
        __syncthreads();

        const _Float16* hbase = hbuf + (size_t)(hb + r0) * F;
#ifdef USE_TDM
        if (wave == 0) tdm_tile_f16(ldsA0, hbase, MT);
#else
        int ra0 = r0 + am0; if (ra0 >= cnt) ra0 = cnt - 1;
        int ra1 = r0 + am1; if (ra1 >= cnt) ra1 = cnt - 1;
        const _Float16* hrow0 = hbuf + (size_t)(hb + ra0) * F + aq;
        const _Float16* hrow1 = hbuf + (size_t)(hb + ra1) * F + aq;
        uint4 av0 = *(const uint4*)(hrow0);
        uint4 av1 = *(const uint4*)(hrow1);
        *(uint4*)&Ast[am0 * SA + aq] = av0;
        *(uint4*)&Ast[am1 * SA + aq] = av1;
#endif

        v8f acc[2][2] = {};
        int stage = 0;
#pragma unroll 1
        for (int it = 0; it < F / 32; ++it) {
#ifdef USE_TDM
            __builtin_amdgcn_s_wait_tensorcnt(0);
#endif
            __syncthreads();
            int kn = (it + 1) * 32;
#ifdef USE_TDM
            if (kn < F && wave == 0)
                tdm_tile_f16(stage ? ldsA0 : ldsA1, hbase + kn, MT);
#else
            if (kn < F) {
                av0 = *(const uint4*)(hrow0 + kn);
                av1 = *(const uint4*)(hrow1 + kn);
            }
#endif
            const _Float16* As = &Ast[stage * MT * SA];
            v16h a0 = load_frag16(&As[(wm * 32 + 0) * SA], SA, lane);
            v16h a1 = load_frag16(&As[(wm * 32 + 16) * SA], SA, lane);
            v16h b0 = load_frag16(&Bp[(wn * 32 + 0) * SBD + it * 32], SBD, lane);
            v16h b1 = load_frag16(&Bp[(wn * 32 + 16) * SBD + it * 32], SBD, lane);
            acc[0][0] = __builtin_amdgcn_wmma_f32_16x16x32_f16(false, a0, false, b0, (short)0, acc[0][0], false, false);
            acc[1][0] = __builtin_amdgcn_wmma_f32_16x16x32_f16(false, a1, false, b0, (short)0, acc[1][0], false, false);
            acc[0][1] = __builtin_amdgcn_wmma_f32_16x16x32_f16(false, a0, false, b1, (short)0, acc[0][1], false, false);
            acc[1][1] = __builtin_amdgcn_wmma_f32_16x16x32_f16(false, a1, false, b1, (short)0, acc[1][1], false, false);
#ifdef USE_TDM
            if (kn < F) stage ^= 1;
#else
            if (kn < F) {
                stage ^= 1;
                _Float16* Asw = &Ast[stage * MT * SA];
                *(uint4*)&Asw[am0 * SA + aq] = av0;
                *(uint4*)&Asw[am1 * SA + aq] = av1;
            }
#endif
        }

#pragma unroll
        for (int ni = 0; ni < 2; ++ni) {
            int gn = n0 + wn * 32 + ni * 16 + ncol;
#pragma unroll
            for (int mi = 0; mi < 2; ++mi)
#pragma unroll
                for (int v = 0; v < 8; ++v) {
                    int gm = wm * 32 + mi * 16 + half8 + v;
                    int r = r0 + gm;
                    if (r < cnt) {
                        size_t o = (size_t)toks[gm] * D + gn;
                        out[o] += wsh[gm] * acc[mi][ni][v];
                    }
                }
        }
    }
}

// ---------------------------------------------------------------------------
extern "C" void kernel_launch(void* const* d_in, const int* in_sizes, int n_in,
                              void* d_out, int out_size, void* d_ws, size_t ws_size,
                              hipStream_t stream) {
    const float* x  = (const float*)d_in[0];   // [T, D]
    const float* gw = (const float*)d_in[1];   // [E, D]
    const float* wg = (const float*)d_in[2];   // [E, F, D]
    const float* wu = (const float*)d_in[3];   // [E, F, D]
    const float* wd = (const float*)d_in[4];   // [E, D, F]

    float* out    = (float*)d_out;             // [T, D]
    float* logits = out + (size_t)T * D;       // [T, E] (second tuple output)

    // workspace (~42.7 MB): rows int[16T], wtsb float[16T], counts int[16],
    // base int[17], hbuf f16[(2T+MT)*F], x16 f16[T*D]
    int*      rows   = (int*)d_ws;
    float*    wtsb   = (float*)((char*)d_ws + (size_t)16 * T * 4);
    int*      counts = (int*)((char*)d_ws + (size_t)32 * T * 4);
    int*      basep  = counts + 16;
    _Float16* hbuf   = (_Float16*)((char*)d_ws + (size_t)32 * T * 4 + 1024);
    _Float16* x16    = hbuf + (size_t)(2 * T + MT) * F;

    k_zero<<<(T * D / 4 + 255) / 256, 256, 0, stream>>>(out, T * D / 4, counts);
    k_cvtx<<<T * D / (256 * 8), 256, 0, stream>>>(x, x16);
    k_router<<<T, 256, 0, stream>>>(x, gw, logits);
    k_sparsemixer<<<T / 256, 256, 0, stream>>>(logits, rows, wtsb, counts);
    k_prefix<<<1, 32, 0, stream>>>(counts, basep);

    size_t smem_gu = (size_t)(2 * 64 * SBG + 2 * MT * SA) * 2 + MT * 4;
    dim3 ggu(F / 64, 16);
    k_gateup<<<ggu, 256, smem_gu, stream>>>(x16, wg, wu, rows, counts, basep, hbuf);

    size_t smem_dn = (size_t)(64 * SBD + 2 * MT * SA) * 2 + MT * 4 + MT * 4;
    dim3 gdn(D / 64, E);
    k_down<<<gdn, 256, smem_dn, stream>>>(wd, rows, wtsb, counts, basep, hbuf, out, 0);
    k_down<<<gdn, 256, smem_dn, stream>>>(wd, rows, wtsb, counts, basep, hbuf, out, 1);
}
